// GPT2_76295799046281
// MI455X (gfx1250) — matrix operand order repaired and checked by reading, not compiled, not run on hardware
//
#include <hip/hip_runtime.h>
#include <hip/hip_bf16.h>
#include <hip/hip_fp16.h>

// Problem constants (from reference): B=2, T=2048, C=1024, H=16, D=64
#define BQ 2
#define TQ 2048
#define CQ 1024
#define HQ 16
#define DQ 64

typedef __attribute__((ext_vector_type(4)))  _Float16 v4h;
typedef __attribute__((ext_vector_type(8)))  _Float16 v8h;
typedef __attribute__((ext_vector_type(16))) _Float16 v16h;
typedef __attribute__((ext_vector_type(8)))  float    v8f;

// ---------------------------------------------------------------------------
// WMMA wrapper: D(f32 16x16) = A(f16 16x32) * B(f16 32x16) + C
// 8-arg form confirmed for gfx1250: (neg_a, A, neg_b, B, c_mod, C, reuse_a, reuse_b)
// ---------------------------------------------------------------------------
__device__ __forceinline__ v8f wmma_f16(v16h a, v16h b, v8f c) {
  return __builtin_amdgcn_wmma_f32_16x16x32_f16(false, a, false, b, (short)0, c,
                                                false, false);
}

__device__ __forceinline__ v16h combine8(v8h lo, v8h hi) {
  v16h r;
#pragma unroll
  for (int i = 0; i < 8; ++i) { r[i] = lo[i]; r[i + 8] = hi[i]; }
  return r;
}

// A-fragment (16x32 f16, row-major [m][k] with leading dim `ld` halfs).
// ISA layout: lane L: m = L%16; halfs 0..7 -> K = kb..kb+7, halfs 8..15 ->
// K = kb+16..kb+23, kb = (L<16) ? 0 : 8.
__device__ __forceinline__ v16h load_a16x32(const _Float16* base, int ld) {
  const int lane = threadIdx.x & 31;
  const int m  = lane & 15;
  const int kb = (lane & 16) ? 8 : 0;
  const _Float16* p = base + m * ld + kb;
  return combine8(*(const v8h*)p, *(const v8h*)(p + 16));
}

// B-fragment (32x16 f16). Source stored as B^T row-major: [n][k], leading dim
// `ld` halfs. ISA layout: lane L: n = L%16; halfs 0..7 -> K = kb..kb+7,
// halfs 8..15 -> K = kb+16..kb+23, kb = (L<16) ? 0 : 8.
__device__ __forceinline__ v16h load_b32x16(const _Float16* base, int ld) {
  const int lane = threadIdx.x & 31;
  const int n  = lane & 15;
  const int kb = (lane & 16) ? 8 : 0;
  const _Float16* p = base + n * ld + kb;
  return combine8(*(const v8h*)p, *(const v8h*)(p + 16));
}

// ---------------------------------------------------------------------------
// Kernel 1: fused QKV projection.
//   out[m, o] = sum_k x[m,k] * W[o,k] + b[o]   (nn.Linear: x @ W.T + b)
// M = B*T = 4096 rows, virtual N = 3*C = 3072 (Wq | Wk | Wv).
// Block tile 128x128, 8 waves (4x2), wave tile 32x64, K-step 32.
// Q,K stored f16 [b,h,t,d]; V stored f16 transposed [b,h,d,t].
// ---------------------------------------------------------------------------
__global__ void __launch_bounds__(256) qkv_gemm_kernel(
    const float* __restrict__ x,
    const float* __restrict__ Wqp, const float* __restrict__ Wkp,
    const float* __restrict__ Wvp,
    const float* __restrict__ bqp, const float* __restrict__ bkp,
    const float* __restrict__ bvp,
    _Float16* __restrict__ q16, _Float16* __restrict__ k16,
    _Float16* __restrict__ v16) {
  __shared__ _Float16 sA[128 * 40];  // [m][k], ld=40 halfs (80B rows)
  __shared__ _Float16 sB[128 * 40];  // [o][k] = B^T, ld=40

  const int tid  = threadIdx.x;
  const int lane = tid & 31;
  const int wave = tid >> 5;
  const int wm = (wave & 3) * 32;   // wave row offset within 128
  const int wn = (wave >> 2) * 64;  // wave col offset within 128
  const int m0 = blockIdx.x * 128;
  const int n0 = blockIdx.y * 128;       // virtual column over [Wq;Wk;Wv]
  const int mat = n0 >> 10;              // 0=Q, 1=K, 2=V (blocks never straddle)
  const int o0  = n0 & 1023;

  const float* W    = (mat == 0) ? Wqp : (mat == 1) ? Wkp : Wvp;
  const float* bias = (mat == 0) ? bqp : (mat == 1) ? bkp : bvp;
  _Float16*    dst  = (mat == 0) ? q16 : (mat == 1) ? k16 : v16;

  v8f acc[2][4];
#pragma unroll
  for (int i = 0; i < 2; ++i)
#pragma unroll
    for (int j = 0; j < 4; ++j) acc[i][j] = (v8f)0.0f;

  for (int k0 = 0; k0 < CQ; k0 += 32) {
    __syncthreads();
    // Stage A (x tile 128x32 f32 -> f16) and B (W tile 128x32 f32 -> f16).
#pragma unroll
    for (int j = 0; j < 4; ++j) {
      const int c   = tid + j * 256;     // 0..1023 : 128 rows x 8 float4-chunks
      const int row = c >> 3;
      const int c4  = c & 7;
      float4 fa = *(const float4*)(x + (size_t)(m0 + row) * CQ + k0 + c4 * 4);
      v4h ha; ha[0] = (_Float16)fa.x; ha[1] = (_Float16)fa.y;
              ha[2] = (_Float16)fa.z; ha[3] = (_Float16)fa.w;
      *(v4h*)(sA + row * 40 + c4 * 4) = ha;
      float4 fb = *(const float4*)(W + (size_t)(o0 + row) * CQ + k0 + c4 * 4);
      v4h hb2; hb2[0] = (_Float16)fb.x; hb2[1] = (_Float16)fb.y;
               hb2[2] = (_Float16)fb.z; hb2[3] = (_Float16)fb.w;
      *(v4h*)(sB + row * 40 + c4 * 4) = hb2;
    }
    __syncthreads();

    v16h a0 = load_a16x32(sA + (wm + 0)  * 40, 40);
    v16h a1 = load_a16x32(sA + (wm + 16) * 40, 40);
#pragma unroll
    for (int nt = 0; nt < 4; ++nt) {
      v16h bf = load_b32x16(sB + (wn + nt * 16) * 40, 40);
      acc[0][nt] = wmma_f16(a0, bf, acc[0][nt]);
      acc[1][nt] = wmma_f16(a1, bf, acc[1][nt]);
    }
  }

  // Epilogue: C-frag layout: lane holds column n = lane%16, rows r + 8*(lane>=16).
  const int nl = lane & 15;
  const int hb = (lane & 16) ? 8 : 0;
#pragma unroll
  for (int mt = 0; mt < 2; ++mt) {
#pragma unroll
    for (int nt = 0; nt < 4; ++nt) {
      const int o  = o0 + wn + nt * 16 + nl;  // column within this matrix
      const float bval = bias[o];
      const int hh = o >> 6;
      const int d  = o & 63;
      const int mrow0 = m0 + wm + mt * 16 + hb;  // 8 consecutive rows, same b
      const int bb = mrow0 >> 11;
      const int t0 = mrow0 & 2047;
      if (mat < 2) {
        // [b,h,t,d]
        _Float16* p = dst + (((size_t)(bb * HQ + hh)) * TQ + t0) * DQ + d;
#pragma unroll
        for (int r = 0; r < 8; ++r)
          p[(size_t)r * DQ] = (_Float16)(acc[mt][nt][r] + bval);
      } else {
        // V transposed: [b,h,d,t] -> 8 contiguous t per lane
        v8h pack;
#pragma unroll
        for (int r = 0; r < 8; ++r) pack[r] = (_Float16)(acc[mt][nt][r] + bval);
        *(v8h*)(dst + ((size_t)(bb * HQ + hh) * DQ + d) * TQ + t0) = pack;
      }
    }
  }
}

// ---------------------------------------------------------------------------
// Kernel 2: causal ALiBi flash attention.
// Grid: (T/128, B*H). Block = 256 threads = 8 waves; wave owns 16 q-rows.
// K-tile = 64 keys; S = Q*K^T via WMMA, online softmax in f32 registers,
// P re-laid out through LDS into A-fragments, O += P*V via WMMA.
// ---------------------------------------------------------------------------
__global__ void __launch_bounds__(256) attn_kernel(
    const _Float16* __restrict__ q16, const _Float16* __restrict__ k16,
    const _Float16* __restrict__ v16, _Float16* __restrict__ y16) {
  __shared__ _Float16 sK[64 * 80];        // [k_row][d], ld=80
  __shared__ _Float16 sV[64 * 80];        // [d][t]  (V already transposed), ld=80
  __shared__ _Float16 sP[8 * 16 * 80];    // per-wave 16x64 P tile, ld=80

  const int tid  = threadIdx.x;
  const int lane = tid & 31;
  const int wave = tid >> 5;
  const int q0 = blockIdx.x * 128;
  const int bh = blockIdx.y;              // b*H + h
  const int h  = bh & (HQ - 1);
  // slopes for H=16: slope_h = 2^(-0.5*(h+1))
  const float slope = exp2f(-0.5f * (float)(h + 1));

  const _Float16* qb  = q16 + (size_t)bh * TQ * DQ;
  const _Float16* kbp = k16 + (size_t)bh * TQ * DQ;
  const _Float16* vbp = v16 + (size_t)bh * DQ * TQ;
  const int qrow0 = q0 + wave * 16;

  // Q fragments (2 x 16x32 over D=64), pre-scaled by 1/sqrt(D) = 0.125 (exact).
  v16h qf[2];
  {
    const int m = lane & 15;
    const int kbo = (lane & 16) ? 8 : 0;
#pragma unroll
    for (int c = 0; c < 2; ++c) {
      const _Float16* p = qb + (size_t)(qrow0 + m) * DQ + c * 32 + kbo;
      v16h f = combine8(*(const v8h*)p, *(const v8h*)(p + 16));
#pragma unroll
      for (int i = 0; i < 16; ++i) f[i] = f[i] * (_Float16)0.125f;
      qf[c] = f;
    }
  }

  v8f oacc[4];
#pragma unroll
  for (int i = 0; i < 4; ++i) oacc[i] = (v8f)0.0f;
  float mrow[8], lrow[8];
#pragma unroll
  for (int r = 0; r < 8; ++r) { mrow[r] = -1e30f; lrow[r] = 0.0f; }

  const int hb = (lane & 16) ? 8 : 0;
  const int nl = lane & 15;
  const int nkt = (q0 >> 6) + 2;  // key tiles covering k <= q0+127

  for (int kt = 0; kt < nkt; ++kt) {
    const int kt0 = kt * 64;
    __syncthreads();
    // Stage K (64x64 halfs) and V^T (64x64 halfs) tiles into LDS.
#pragma unroll
    for (int j = 0; j < 2; ++j) {
      const int c = tid + j * 256;  // 0..511 : 64 rows x 8 v8h-chunks
      const int row = c >> 3;
      const int c8  = c & 7;
      *(v8h*)(sK + row * 80 + c8 * 8) =
          *(const v8h*)(kbp + (size_t)(kt0 + row) * DQ + c8 * 8);
      *(v8h*)(sV + row * 80 + c8 * 8) =
          *(const v8h*)(vbp + (size_t)row * TQ + kt0 + c8 * 8);
    }
    __syncthreads();

    // S = (Q/8) * K^T   (16 q-rows x 64 keys, 4 n-tiles)
    v8f s[4];
#pragma unroll
    for (int i = 0; i < 4; ++i) s[i] = (v8f)0.0f;
#pragma unroll
    for (int c = 0; c < 2; ++c) {
#pragma unroll
      for (int nt = 0; nt < 4; ++nt) {
        v16h bf = load_b32x16(sK + nt * 16 * 80 + c * 32, 80);
        s[nt] = wmma_f16(qf[c], bf, s[nt]);
      }
    }

    // ALiBi bias + causal mask + online softmax (per q-row).
#pragma unroll
    for (int r = 0; r < 8; ++r) {
      const int qg = qrow0 + hb + r;
      float rm = -1e30f;
#pragma unroll
      for (int nt = 0; nt < 4; ++nt) {
        const int kg = kt0 + nt * 16 + nl;
        float v = s[nt][r] + slope * (float)(kg - qg);  // = att - slope*(q-k)
        v = (kg <= qg) ? v : -1e30f;
        s[nt][r] = v;
        rm = fmaxf(rm, v);
      }
#pragma unroll
      for (int off = 1; off < 16; off <<= 1)
        rm = fmaxf(rm, __shfl_xor(rm, off, 32));
      const float mn = fmaxf(mrow[r], rm);
      const float alpha = __expf(mrow[r] - mn);
      float rs = 0.0f;
#pragma unroll
      for (int nt = 0; nt < 4; ++nt) {
        const float p = __expf(s[nt][r] - mn);
        s[nt][r] = p;
        rs += p;
      }
#pragma unroll
      for (int off = 1; off < 16; off <<= 1) rs += __shfl_xor(rs, off, 32);
      lrow[r] = lrow[r] * alpha + rs;
      mrow[r] = mn;
#pragma unroll
      for (int dt = 0; dt < 4; ++dt) oacc[dt][r] *= alpha;
    }

    // P (C-frag layout) -> LDS [q][k] so it can be re-read as A-fragments.
    _Float16* myP = sP + wave * 16 * 80;
#pragma unroll
    for (int nt = 0; nt < 4; ++nt)
#pragma unroll
      for (int r = 0; r < 8; ++r)
        myP[(hb + r) * 80 + nt * 16 + nl] = (_Float16)s[nt][r];
    __syncthreads();

    // O += P * V   (V^T rows give contiguous-K B-fragments)
#pragma unroll
    for (int c = 0; c < 2; ++c) {
      v16h pf = load_a16x32(myP + c * 32, 80);
#pragma unroll
      for (int dt = 0; dt < 4; ++dt) {
        v16h vf = load_b32x16(sV + dt * 16 * 80 + c * 32, 80);
        oacc[dt] = wmma_f16(pf, vf, oacc[dt]);
      }
    }
  }

  // Normalize and store y (f16, [b, t, c] with c = h*64 + d).
  _Float16* yb = y16 + ((size_t)(bh >> 4) * TQ) * CQ + h * DQ;
#pragma unroll
  for (int r = 0; r < 8; ++r) {
    const float inv = 1.0f / lrow[r];
    const int t = qrow0 + hb + r;
#pragma unroll
    for (int dt = 0; dt < 4; ++dt)
      yb[(size_t)t * CQ + dt * 16 + nl] = (_Float16)(oacc[dt][r] * inv);
  }
}

// ---------------------------------------------------------------------------
// Kernel 3: output projection.  out[m,o] = sum_k y16[m,k]*Wo[o,k] + bo[o], f32.
// Same tiling as kernel 1; A already f16 in workspace.
// ---------------------------------------------------------------------------
__global__ void __launch_bounds__(256) out_gemm_kernel(
    const _Float16* __restrict__ y16, const float* __restrict__ Wo,
    const float* __restrict__ bo, float* __restrict__ out) {
  __shared__ _Float16 sA[128 * 40];
  __shared__ _Float16 sB[128 * 40];

  const int tid  = threadIdx.x;
  const int lane = tid & 31;
  const int wave = tid >> 5;
  const int wm = (wave & 3) * 32;
  const int wn = (wave >> 2) * 64;
  const int m0 = blockIdx.x * 128;
  const int n0 = blockIdx.y * 128;

  v8f acc[2][4];
#pragma unroll
  for (int i = 0; i < 2; ++i)
#pragma unroll
    for (int j = 0; j < 4; ++j) acc[i][j] = (v8f)0.0f;

  for (int k0 = 0; k0 < CQ; k0 += 32) {
    __syncthreads();
    // A tile (f16 copy): 128 rows x 4 v8h chunks = 512 chunks.
#pragma unroll
    for (int j = 0; j < 2; ++j) {
      const int c = tid + j * 256;
      const int row = c >> 2;
      const int c4  = c & 3;
      *(v8h*)(sA + row * 40 + c4 * 8) =
          *(const v8h*)(y16 + (size_t)(m0 + row) * CQ + k0 + c4 * 8);
    }
    // B tile (Wo f32 -> f16): 128 rows x 8 float4 chunks = 1024 chunks.
#pragma unroll
    for (int j = 0; j < 4; ++j) {
      const int c = tid + j * 256;
      const int row = c >> 3;
      const int c4  = c & 7;
      float4 fb = *(const float4*)(Wo + (size_t)(n0 + row) * CQ + k0 + c4 * 4);
      v4h hb2; hb2[0] = (_Float16)fb.x; hb2[1] = (_Float16)fb.y;
               hb2[2] = (_Float16)fb.z; hb2[3] = (_Float16)fb.w;
      *(v4h*)(sB + row * 40 + c4 * 4) = hb2;
    }
    __syncthreads();

    v16h a0 = load_a16x32(sA + (wm + 0)  * 40, 40);
    v16h a1 = load_a16x32(sA + (wm + 16) * 40, 40);
#pragma unroll
    for (int nt = 0; nt < 4; ++nt) {
      v16h bf = load_b32x16(sB + (wn + nt * 16) * 40, 40);
      acc[0][nt] = wmma_f16(a0, bf, acc[0][nt]);
      acc[1][nt] = wmma_f16(a1, bf, acc[1][nt]);
    }
  }

  const int nl = lane & 15;
  const int hb = (lane & 16) ? 8 : 0;
#pragma unroll
  for (int mt = 0; mt < 2; ++mt) {
#pragma unroll
    for (int nt = 0; nt < 4; ++nt) {
      const int o = n0 + wn + nt * 16 + nl;
      const float bval = bo[o];
      const int mrow0 = m0 + wm + mt * 16 + hb;
      float* p = out + (size_t)mrow0 * CQ + o;
#pragma unroll
      for (int r = 0; r < 8; ++r)
        p[(size_t)r * CQ] = acc[mt][nt][r] + bval;
    }
  }
}

// ---------------------------------------------------------------------------
// Launcher. Inputs: x, Wq, bq, Wk, bk, Wv, bv, Wo, bo (all f32).
// Workspace layout (f16): Q | K | V^T | y  = 4 x 4M halfs = 32 MB total.
// ---------------------------------------------------------------------------
extern "C" void kernel_launch(void* const* d_in, const int* in_sizes, int n_in,
                              void* d_out, int out_size, void* d_ws,
                              size_t ws_size, hipStream_t stream) {
  const float* x  = (const float*)d_in[0];
  const float* Wq = (const float*)d_in[1];
  const float* bq = (const float*)d_in[2];
  const float* Wk = (const float*)d_in[3];
  const float* bk = (const float*)d_in[4];
  const float* Wv = (const float*)d_in[5];
  const float* bv = (const float*)d_in[6];
  const float* Wo = (const float*)d_in[7];
  const float* bo = (const float*)d_in[8];
  float* out = (float*)d_out;

  const size_t elems = (size_t)BQ * HQ * TQ * DQ;  // 4,194,304 halfs each
  _Float16* q16 = (_Float16*)d_ws;
  _Float16* k16 = q16 + elems;
  _Float16* v16 = k16 + elems;
  _Float16* y16 = v16 + elems;

  // 1) QKV projections: grid = (M/128, 3N/128) = (32, 24)
  qkv_gemm_kernel<<<dim3(32, 24), 256, 0, stream>>>(
      x, Wq, Wk, Wv, bq, bk, bv, q16, k16, v16);

  // 2) Attention: grid = (T/128, B*H) = (16, 32)
  attn_kernel<<<dim3(16, 32), 256, 0, stream>>>(q16, k16, v16, y16);

  // 3) Output projection: grid = (M/128, N/128) = (32, 8)
  out_gemm_kernel<<<dim3(32, 8), 256, 0, stream>>>(y16, Wo, bo, out);
}
